// GraphEncoder_8297876816594
// MI455X (gfx1250) — matrix operand order repaired and checked by reading, not compiled, not run on hardware
//
#include <hip/hip_runtime.h>

#define N_NODES_C 100000
#define N_EDGES_C 6400000
#define N_GRAPHS_C 512
#define IN_DIM_C 4
#define HID_C 64
#define OUT_DIM_C 128

typedef __attribute__((ext_vector_type(16))) _Float16 v16h;
typedef __attribute__((ext_vector_type(8)))  float    v8f;

// A-fragment k order chosen here (16x32 f16, not AMD's table order): lane l holds row m=l&15, hi=l>>4.
// half j: k = hi*16 + j (contiguous K per lane-half, matches B)
__device__ __forceinline__ int wmma_kA(int j, int hi) {
    return (hi << 4) | j;
}
// B-fragment k order chosen here (32x16 f16): lane l holds col n=l&15; halves j=0..15 hold
// K = hi*16 + j  (lanes 0-15: K=0-15, lanes 16-31: K=16-31) -> K contiguous.

// ---------------------------------------------------------------- zero ws
__global__ __launch_bounds__(256) void zero_kernel(float* __restrict__ p, int n) {
    int i = blockIdx.x * 256 + threadIdx.x;
    if (i < n) p[i] = 0.0f;
}

// ------------------------------------------------- edge scatter (GIN agg)
__global__ __launch_bounds__(256) void edge_agg_kernel(
    const float4* __restrict__ x, const long long* __restrict__ src,
    const long long* __restrict__ dst, float* __restrict__ agg) {
    int e = blockIdx.x * 256 + threadIdx.x;
    if (e >= N_EDGES_C) return;
    // prefetch the streaming index arrays ahead (speculative: OOB is dropped)
    __builtin_prefetch(&src[e + 4096], 0, 1);
    __builtin_prefetch(&dst[e + 4096], 0, 1);
    int s = (int)src[e];
    int d = (int)dst[e];
    float4 v = x[s];
    atomicAdd(&agg[d * 4 + 0], v.x);
    atomicAdd(&agg[d * 4 + 1], v.y);
    atomicAdd(&agg[d * 4 + 2], v.z);
    atomicAdd(&agg[d * 4 + 3], v.w);
}

// ---------------------------------------------------------- graph counts
__global__ __launch_bounds__(256) void count_kernel(
    const long long* __restrict__ batch_ids, float* __restrict__ counts) {
    int n = blockIdx.x * 256 + threadIdx.x;
    if (n >= N_NODES_C) return;
    atomicAdd(&counts[(int)batch_ids[n]], 1.0f);
}

// ---- fused: h = relu((x+agg)@W1+b1); h2 = h@W2+b2 (WMMA); pool-sum atomics
__global__ __launch_bounds__(256) void gin_mlp_pool_kernel(
    const float4* __restrict__ x, const float4* __restrict__ agg,
    const long long* __restrict__ batch_ids,
    const float* __restrict__ W1, const float* __restrict__ b1,
    const float* __restrict__ W2, const float* __restrict__ b2,
    float* __restrict__ sums) {
    __shared__ float sW1[IN_DIM_C * HID_C];
    __shared__ float sb1[HID_C];
    __shared__ float sb2[HID_C];
    // W2 transposed + pre-converted to f16: sW2t[n*64 + k]. B fragment for
    // (col n, lane-half hi) is 16 contiguous halves -> one 32B vector LDS load.
    __shared__ __align__(32) _Float16 sW2t[HID_C * HID_C];

    for (int i = threadIdx.x; i < IN_DIM_C * HID_C; i += 256) sW1[i] = W1[i];
    for (int i = threadIdx.x; i < HID_C * HID_C; i += 256) {
        int k = i >> 6, n = i & 63;            // W2 is [k][n] row-major
        sW2t[n * HID_C + k] = (_Float16)W2[i];
    }
    if (threadIdx.x < HID_C) {
        sb1[threadIdx.x] = b1[threadIdx.x];
        sb2[threadIdx.x] = b2[threadIdx.x];
    }
    __syncthreads();

    const int lane = threadIdx.x & 31;
    const int wave = threadIdx.x >> 5;
    const int m    = lane & 15;
    const int hi   = lane >> 4;
    const int nodeBase = (blockIdx.x * 8 + wave) * 16;
    if (nodeBase >= N_NODES_C) return;          // wave-uniform exit

    // ---- A fragments (16 nodes x K=64, f16): compute h1 on the fly
    int nodeA = nodeBase + m;
    int nodeC = nodeA < N_NODES_C ? nodeA : N_NODES_C - 1;  // clamp (dups ok)
    float4 xv = x[nodeC];
    float4 av = agg[nodeC];
    float xa0 = xv.x + av.x, xa1 = xv.y + av.y;
    float xa2 = xv.z + av.z, xa3 = xv.w + av.w;

    v16h afrag0, afrag1;
#pragma unroll
    for (int j = 0; j < 16; ++j) {
        int k0 = wmma_kA(j, hi);
        int k1 = k0 + 32;
        float s0 = sb1[k0] + xa0 * sW1[k0] + xa1 * sW1[HID_C + k0] +
                   xa2 * sW1[2 * HID_C + k0] + xa3 * sW1[3 * HID_C + k0];
        float s1 = sb1[k1] + xa0 * sW1[k1] + xa1 * sW1[HID_C + k1] +
                   xa2 * sW1[2 * HID_C + k1] + xa3 * sW1[3 * HID_C + k1];
        afrag0[j] = (_Float16)(s0 > 0.0f ? s0 : 0.0f);
        afrag1[j] = (_Float16)(s1 > 0.0f ? s1 : 0.0f);
    }

    // graph ids for the 8 rows this lane stores (C/D layout: m = r + hi*8).
    // Out-of-range rows: clamp index and later add 0.0f (numeric no-op) so the
    // store path is completely branch-free (no exec-mask churn).
    int   grow[8];
    float wrow[8];
#pragma unroll
    for (int r = 0; r < 8; ++r) {
        int  nr = nodeBase + r + hi * 8;
        bool v  = nr < N_NODES_C;
        wrow[r] = v ? 1.0f : 0.0f;
        grow[r] = (int)batch_ids[v ? nr : (N_NODES_C - 1)];
    }

    const v16h* W2v = (const v16h*)sW2t;  // 16-half (32B) vector view

    // ---- 4 column tiles of the 64-wide output, K=64 -> 2 WMMAs per tile
#pragma unroll
    for (int t = 0; t < 4; ++t) {
        int n = t * 16 + m;
        v16h bfrag0 = W2v[n * 4 + hi];        // K = hi*16 .. hi*16+15
        v16h bfrag1 = W2v[n * 4 + 2 + hi];    // K = 32+hi*16 .. 32+hi*16+15
        float bias = sb2[n];
        v8f acc = {bias, bias, bias, bias, bias, bias, bias, bias};
        acc = __builtin_amdgcn_wmma_f32_16x16x32_f16(false, afrag0, false, bfrag0,
                                                     (short)0, acc, false, false);
        acc = __builtin_amdgcn_wmma_f32_16x16x32_f16(false, afrag1, false, bfrag1,
                                                     (short)0, acc, false, false);
#pragma unroll
        for (int r = 0; r < 8; ++r)
            atomicAdd(&sums[grow[r] * HID_C + n], acc[r] * wrow[r]);
    }
}

// ---- head: out = (sums/max(counts,1)) @ Wlin + blin   [512 x 128], WMMA
__global__ __launch_bounds__(256) void final_gemm_kernel(
    const float* __restrict__ sums, const float* __restrict__ counts,
    const float* __restrict__ Wlin, const float* __restrict__ blin,
    float* __restrict__ out) {
    const int lane = threadIdx.x & 31;
    const int wave = threadIdx.x >> 5;  // column tile 0..7 (128 cols)
    const int m    = lane & 15;
    const int hi   = lane >> 4;
    const int rowBase = blockIdx.x * 16;  // 32 blocks cover 512 graphs
    int row = rowBase + m;

    float c   = counts[row];
    float inv = 1.0f / (c > 1.0f ? c : 1.0f);

    v16h a0, a1;
#pragma unroll
    for (int j = 0; j < 16; ++j) {
        int k0 = wmma_kA(j, hi);
        a0[j] = (_Float16)(sums[row * HID_C + k0] * inv);
        a1[j] = (_Float16)(sums[row * HID_C + k0 + 32] * inv);
    }

    int n = wave * 16 + m;
    v16h bf0, bf1;
#pragma unroll
    for (int j = 0; j < 16; ++j) {
        int kb = hi * 16 + j;  // B layout: contiguous K per lane-half
        bf0[j] = (_Float16)Wlin[kb * OUT_DIM_C + n];
        bf1[j] = (_Float16)Wlin[(kb + 32) * OUT_DIM_C + n];
    }
    float bias = blin[n];
    v8f acc = {bias, bias, bias, bias, bias, bias, bias, bias};
    acc = __builtin_amdgcn_wmma_f32_16x16x32_f16(false, a0, false, bf0,
                                                 (short)0, acc, false, false);
    acc = __builtin_amdgcn_wmma_f32_16x16x32_f16(false, a1, false, bf1,
                                                 (short)0, acc, false, false);
#pragma unroll
    for (int r = 0; r < 8; ++r) {
        out[(rowBase + r + hi * 8) * OUT_DIM_C + n] = acc[r];
    }
}

extern "C" void kernel_launch(void* const* d_in, const int* in_sizes, int n_in,
                              void* d_out, int out_size, void* d_ws, size_t ws_size,
                              hipStream_t stream) {
    (void)in_sizes; (void)n_in; (void)out_size; (void)ws_size;
    const float*     x     = (const float*)d_in[0];
    const long long* ei    = (const long long*)d_in[1];  // int64 [2, E]
    const long long* batch = (const long long*)d_in[2];  // int64 [N]
    const float*     W1    = (const float*)d_in[3];
    const float*     b1    = (const float*)d_in[4];
    const float*     W2    = (const float*)d_in[5];
    const float*     b2    = (const float*)d_in[6];
    const float*     Wlin  = (const float*)d_in[7];
    const float*     blin  = (const float*)d_in[8];
    float*           out   = (float*)d_out;

    // workspace layout (floats): agg[N*4] | sums[G*64] | counts[G]
    float* agg    = (float*)d_ws;
    float* sums   = agg + (size_t)N_NODES_C * IN_DIM_C;
    float* counts = sums + (size_t)N_GRAPHS_C * HID_C;

    int zn = N_NODES_C * IN_DIM_C + N_GRAPHS_C * HID_C + N_GRAPHS_C;
    zero_kernel<<<(zn + 255) / 256, 256, 0, stream>>>(agg, zn);
    edge_agg_kernel<<<(N_EDGES_C + 255) / 256, 256, 0, stream>>>(
        (const float4*)x, ei, ei + N_EDGES_C, agg);
    count_kernel<<<(N_NODES_C + 255) / 256, 256, 0, stream>>>(batch, counts);
    gin_mlp_pool_kernel<<<(N_NODES_C + 127) / 128, 256, 0, stream>>>(
        (const float4*)x, (const float4*)agg, batch, W1, b1, W2, b2, sums);
    final_gemm_kernel<<<N_GRAPHS_C / 16, 256, 0, stream>>>(sums, counts, Wlin, blin, out);
}
